// MultiModalMoERec_74045236183782
// MI455X (gfx1250) — matrix operand order repaired and checked by reading, not compiled, not run on hardware
//
#include <hip/hip_runtime.h>

// ---------------------------------------------------------------------------
// MI455X (gfx1250) implementation. All GEMMs via v_wmma_f32_16x16x32_bf16
// (fp32 accumulate), elementwise math in fp32. wave32 everywhere.
// ---------------------------------------------------------------------------

typedef __bf16 bf16;
typedef bf16  v16bf __attribute__((ext_vector_type(16)));
typedef float v8f   __attribute__((ext_vector_type(8)));
typedef unsigned int v4u __attribute__((ext_vector_type(4)));

union FragU { v16bf v; v4u q[2]; };

__device__ __forceinline__ v8f zerov8() {
  v8f z; for (int i = 0; i < 8; i++) z[i] = 0.0f; return z;
}

__device__ __forceinline__ v8f wmma_bf16(v16bf a, v16bf b, v8f c) {
  // (neg_a, A, neg_b, B, c_mod, C, reuse_a, reuse_b)
  return __builtin_amdgcn_wmma_f32_16x16x32_bf16(false, a, false, b, (short)0, c,
                                                 false, false);
}

// A-operand fragment: 16x32 bf16, row r = lane%16 of a K-contiguous buffer.
// ISA A packing: half 0 -> K 0..7 & 16..23, half 1 -> K 8..15 & 24..31.
__device__ __forceinline__ v16bf load_a_frag(const bf16* base, int pitch) {
  const int lane = threadIdx.x & 31;
  const int r = lane & 15, hf = lane >> 4;
  const v4u* p = (const v4u*)(base + r * pitch);
  FragU f;
  f.q[0] = p[hf];       // K = hf*8 .. hf*8+7
  f.q[1] = p[2 + hf];   // K = 16+hf*8 .. 16+hf*8+7
  return f.v;
}

// B-operand fragment: 32x16 bf16 taken from an [N][K] row-major buffer
// (element B[k][n] = buf[n][k]). ISA B packing: half 0 -> K 0..7 & 16..23, half 1 -> K 8..15 & 24..31.
__device__ __forceinline__ v16bf load_b_frag(const bf16* base, int pitch) {
  const int lane = threadIdx.x & 31;
  const int r = lane & 15, hf = lane >> 4;
  const v4u* p = (const v4u*)(base + r * pitch);
  FragU f;
  f.q[0] = p[hf];
  f.q[1] = p[2 + hf];
  return f.v;
}

// ---------------------------------------------------------------------------
// K0: weight converts
// ---------------------------------------------------------------------------
__global__ void cvt_f32_bf16(const float* __restrict__ src, bf16* __restrict__ dst, int n) {
  int i = blockIdx.x * 256 + threadIdx.x;
  if (i < n) dst[i] = (bf16)src[i];
}

__global__ void cvt_wr2_pad(const float* __restrict__ src, bf16* __restrict__ dst) {
  int i = blockIdx.x * 256 + threadIdx.x;           // 16*128 = 2048 elems
  if (i < 2048) {
    int r = i >> 7, c = i & 127;
    dst[i] = (bf16)((r < 8) ? src[r * 128 + c] : 0.0f);
  }
}

// ---------------------------------------------------------------------------
// K1a: y[32768 x 768] = x[32768 x 512] @ W^T   (W is [768][512] bf16)
// Block 256 thr = 8 waves (2 m-groups x 4 n-groups), tile 128x128, K-step 32.
// A staged through LDS (fp32->bf16), B fragments direct from L2-resident bf16 W.
// ---------------------------------------------------------------------------
__global__ void __launch_bounds__(256) k1a_gemm(const float* __restrict__ x,
                                                const bf16* __restrict__ Wbf,
                                                float* __restrict__ y) {
  constexpr int PA = 40;                            // 80B pitch: 16B aligned, bank-friendly
  __shared__ alignas(16) bf16 As[128 * PA];
  const int m0 = blockIdx.x * 128;
  const int n0 = blockIdx.y * 128;
  const int tid = threadIdx.x;
  const int w = tid >> 5;
  const int mw = w & 1, nw = w >> 1;
  const int lane = tid & 31, nn = lane & 15, hf = lane >> 4;

  v8f acc[4][2];
  for (int t = 0; t < 4; t++) for (int u = 0; u < 2; u++) acc[t][u] = zerov8();

  for (int k0 = 0; k0 < 512; k0 += 32) {
    __syncthreads();
    for (int i = 0; i < 16; i++) {
      int e = tid + 256 * i;
      int r = e >> 5, c = e & 31;
      As[r * PA + c] = (bf16)x[(m0 + r) * 512 + k0 + c];
    }
    if (k0 + 32 < 512) {                             // gfx1250 global_prefetch of next slab
      int r = tid >> 1, c = (tid & 1) * 16;
      __builtin_prefetch(&x[(m0 + r) * 512 + k0 + 32 + c], 0, 1);
    }
    __syncthreads();
    const bf16* bbase = Wbf + (n0 + nw * 32) * 512 + k0;
    v16bf bf0 = load_b_frag(bbase, 512);
    v16bf bf1 = load_b_frag(bbase + 16 * 512, 512);
    for (int t = 0; t < 4; t++) {
      v16bf af = load_a_frag(&As[(mw * 64 + t * 16) * PA], PA);
      acc[t][0] = wmma_bf16(af, bf0, acc[t][0]);
      acc[t][1] = wmma_bf16(af, bf1, acc[t][1]);
    }
  }
  for (int t = 0; t < 4; t++)
    for (int u = 0; u < 2; u++)
      for (int j = 0; j < 8; j++) {
        int row = m0 + mw * 64 + t * 16 + hf * 8 + j;
        int col = n0 + nw * 32 + u * 16 + nn;
        y[row * 768 + col] = acc[t][u][j];
      }
}

// ---------------------------------------------------------------------------
// K1b: per-row bias + LayerNorm(768) + l2norm + HD^-0.5 pre-scale,
// scattered to bf16 [H][S][B][HD] so K2 tiles are contiguous.
// One wave per row; 8 rows per 256-thread block.
// ---------------------------------------------------------------------------
__global__ void __launch_bounds__(256) k1b_ln(const float* __restrict__ yimg,
                                              const float* __restrict__ ytxt,
                                              const float* __restrict__ bias_i,
                                              const float* __restrict__ g_i,
                                              const float* __restrict__ be_i,
                                              const float* __restrict__ bias_t,
                                              const float* __restrict__ g_t,
                                              const float* __restrict__ be_t,
                                              bf16* __restrict__ Aimg,
                                              bf16* __restrict__ Atxt) {
  const int w = threadIdx.x >> 5, lane = threadIdx.x & 31;
  const int row = blockIdx.x * 8 + w;               // 0..65535
  const int sel = row >= 32768;
  const int m = row & 32767;
  const float* yr = (sel ? ytxt : yimg) + (size_t)m * 768;
  const float* bias = sel ? bias_t : bias_i;
  const float* g = sel ? g_t : g_i;
  const float* be = sel ? be_t : be_i;
  bf16* A = sel ? Atxt : Aimg;

  float v[24];
  float s1 = 0.f, s2 = 0.f;
  for (int i = 0; i < 24; i++) {
    int n = lane + 32 * i;
    float t = yr[n] + bias[n];
    v[i] = t; s1 += t; s2 += t * t;
  }
  for (int msk = 16; msk >= 1; msk >>= 1) {
    s1 += __shfl_xor(s1, msk, 32);
    s2 += __shfl_xor(s2, msk, 32);
  }
  float mu = s1 * (1.0f / 768.0f);
  float rs = rsqrtf(s2 * (1.0f / 768.0f) - mu * mu + 1e-5f);
  float nsq = 0.f;
  for (int i = 0; i < 24; i++) {
    int n = lane + 32 * i;
    float t = (v[i] - mu) * rs * g[n] + be[n];
    v[i] = t; nsq += t * t;
  }
  for (int msk = 16; msk >= 1; msk >>= 1) nsq += __shfl_xor(nsq, msk, 32);
  float inv = 0.125f / fmaxf(sqrtf(nsq), 1e-12f);   // l2norm * HD^-0.5
  const int b = m >> 9, s = m & 511;
  for (int i = 0; i < 24; i++) {
    int n = lane + 32 * i;
    int h = n >> 6, d = n & 63;
    A[(((h * 512) + s) * 64 + b) * 64 + d] = (bf16)(v[i] * inv);
  }
}

// ---------------------------------------------------------------------------
// K2: fully fused batch-axis attention per (h, s, dir):
// Q/K/V proj -> scores -> softmax -> ctx -> out proj + bias + residual.
// 128 threads = 4 waves; wave w owns rows 16w..16w+15 of every 64x64 GEMM.
// ---------------------------------------------------------------------------
__device__ __forceinline__ void proj64(const bf16* __restrict__ A,
                                       const bf16* __restrict__ W,
                                       const float* __restrict__ bias,
                                       bf16* dst, bool transpose,
                                       int w, int nn, int hf) {
  v8f acc[4];
  for (int u = 0; u < 4; u++) acc[u] = zerov8();
  for (int k0 = 0; k0 < 64; k0 += 32) {
    v16bf af = load_a_frag(A + (w * 16) * 64 + k0, 64);
    for (int u = 0; u < 4; u++)
      acc[u] = wmma_bf16(af, load_b_frag(W + (u * 16) * 64 + k0, 64), acc[u]);
  }
  for (int u = 0; u < 4; u++)
    for (int j = 0; j < 8; j++) {
      int mm = w * 16 + hf * 8 + j, nc = u * 16 + nn;
      float val = acc[u][j] + bias[nc];
      if (transpose) dst[nc * 72 + mm] = (bf16)val;   // V stored transposed
      else           dst[mm * 72 + nc] = (bf16)val;
    }
}

__global__ void __launch_bounds__(128) k2_attn(
    const bf16* __restrict__ Aimg, const bf16* __restrict__ Atxt,
    const bf16* __restrict__ headsW,
    const float* __restrict__ bq0, const float* __restrict__ bk0,
    const float* __restrict__ bv0, const float* __restrict__ bo0,
    const float* __restrict__ bq1, const float* __restrict__ bk1,
    const float* __restrict__ bv1, const float* __restrict__ bo1,
    float* __restrict__ comb) {
  constexpr int P = 72;                              // 144B pitch, 16B aligned
  __shared__ alignas(16) bf16 Qs [64 * P];
  __shared__ alignas(16) bf16 Ks [64 * P];
  __shared__ alignas(16) bf16 VTs[64 * P];
  __shared__ alignas(16) bf16 Ps_[64 * P];
  __shared__ alignas(16) bf16 Cs [64 * P];
  __shared__ float Ss[64 * 68];

  const int blk = blockIdx.x;                        // h*512 + s
  const int dir = blockIdx.y;
  const int h = blk >> 9;
  const int tid = threadIdx.x;
  const int w = tid >> 5, lane = tid & 31;
  const int nn = lane & 15, hf = lane >> 4;

  const bf16* Xq  = (dir ? Atxt : Aimg) + (size_t)blk * 4096;
  const bf16* Xkv = (dir ? Aimg : Atxt) + (size_t)blk * 4096;
  const bf16* Wq = headsW + ((size_t)dir * 4 + 0) * 49152 + h * 4096;
  const bf16* Wk = headsW + ((size_t)dir * 4 + 1) * 49152 + h * 4096;
  const bf16* Wv = headsW + ((size_t)dir * 4 + 2) * 49152 + h * 4096;
  const bf16* Wo = headsW + ((size_t)dir * 4 + 3) * 49152 + h * 4096;
  const float* bq = (dir ? bq1 : bq0) + h * 64;
  const float* bk = (dir ? bk1 : bk0) + h * 64;
  const float* bv = (dir ? bv1 : bv0) + h * 64;
  const float* bo = (dir ? bo1 : bo0) + h * 64;

  proj64(Xq,  Wq, bq, Qs,  false, w, nn, hf);
  proj64(Xkv, Wk, bk, Ks,  false, w, nn, hf);
  proj64(Xkv, Wv, bv, VTs, true,  w, nn, hf);
  __syncthreads();

  {                                                  // scores = Q K^T * HD^-0.5
    v8f acc[4];
    for (int u = 0; u < 4; u++) acc[u] = zerov8();
    for (int k0 = 0; k0 < 64; k0 += 32) {
      v16bf af = load_a_frag(&Qs[(w * 16) * P + k0], P);
      for (int u = 0; u < 4; u++)
        acc[u] = wmma_bf16(af, load_b_frag(&Ks[(u * 16) * P + k0], P), acc[u]);
    }
    for (int u = 0; u < 4; u++)
      for (int j = 0; j < 8; j++)
        Ss[(w * 16 + hf * 8 + j) * 68 + u * 16 + nn] = acc[u][j] * 0.125f;
  }
  __syncthreads();

  if (tid < 64) {                                    // softmax over key-batch axis
    float mx = -3.4e38f;
    for (int c = 0; c < 64; c++) mx = fmaxf(mx, Ss[tid * 68 + c]);
    float sum = 0.f;
    for (int c = 0; c < 64; c++) sum += expf(Ss[tid * 68 + c] - mx);
    float inv = 1.0f / sum;
    for (int c = 0; c < 64; c++)
      Ps_[tid * P + c] = (bf16)(expf(Ss[tid * 68 + c] - mx) * inv);
  }
  __syncthreads();

  {                                                  // ctx = P V   (B-op from V^T rows)
    v8f acc[4];
    for (int u = 0; u < 4; u++) acc[u] = zerov8();
    for (int k0 = 0; k0 < 64; k0 += 32) {
      v16bf af = load_a_frag(&Ps_[(w * 16) * P + k0], P);
      for (int u = 0; u < 4; u++)
        acc[u] = wmma_bf16(af, load_b_frag(&VTs[(u * 16) * P + k0], P), acc[u]);
    }
    for (int u = 0; u < 4; u++)
      for (int j = 0; j < 8; j++)
        Cs[(w * 16 + hf * 8 + j) * P + u * 16 + nn] = (bf16)acc[u][j];
  }
  __syncthreads();

  {                                                  // out = ctx Wo^T + bo + residual
    v8f acc[4];
    for (int u = 0; u < 4; u++) acc[u] = zerov8();
    for (int k0 = 0; k0 < 64; k0 += 32) {
      v16bf af = load_a_frag(&Cs[(w * 16) * P + k0], P);
      for (int u = 0; u < 4; u++)
        acc[u] = wmma_bf16(af, load_b_frag(Wo + (u * 16) * 64 + k0, 64), acc[u]);
    }
    for (int u = 0; u < 4; u++)
      for (int j = 0; j < 8; j++) {
        int mm = w * 16 + hf * 8 + j, nc = u * 16 + nn;
        float val = acc[u][j] + bo[nc] + (float)Xq[mm * 64 + nc];
        comb[((size_t)(blk * 64 + mm)) * 128 + dir * 64 + nc] = val;
      }
  }
}

// ---------------------------------------------------------------------------
// K3: per-64-token block (one (h,s) slice): head-LN -> nan_to_num -> l2norm ->
// LN -> WMMA 64x128x128 -> exact GELU -> LN (shuffle reductions on accum
// fragments) -> WMMA 64x16x128 (W_r2 zero-padded) -> log_softmax/T -> gates.
// ---------------------------------------------------------------------------
__global__ void __launch_bounds__(128) k3_router(
    const float* __restrict__ comb,
    const float* __restrict__ g_hn, const float* __restrict__ b_hn,
    const float* __restrict__ g_r1, const float* __restrict__ b_r1,
    const bf16*  __restrict__ Wr1,  const float* __restrict__ bb_r1,
    const float* __restrict__ g_r2, const float* __restrict__ b_r2,
    const bf16*  __restrict__ Wr2p, const float* __restrict__ bb_r2,
    float* __restrict__ out) {
  constexpr int P = 136;                             // 272B pitch, 16B aligned
  __shared__ alignas(16) bf16 X1[64 * P];
  __shared__ alignas(16) bf16 X2[64 * P];
  const int blk = blockIdx.x;                        // h*512 + s
  const int h = blk >> 9, s = blk & 511;
  const int tid = threadIdx.x;

  {                                                  // elementwise front end: 2 thr/row
    const int r = tid >> 1, half = tid & 1;
    const float* row = comb + ((size_t)blk * 64 + r) * 128 + half * 64;
    float v[64];
    float s1 = 0.f, s2 = 0.f;
    for (int i = 0; i < 64; i++) { float t = row[i]; v[i] = t; s1 += t; s2 += t * t; }
    s1 += __shfl_xor(s1, 1, 32); s2 += __shfl_xor(s2, 1, 32);
    float mu = s1 * (1.f / 128.f);
    float rs = rsqrtf(s2 * (1.f / 128.f) - mu * mu + 1e-5f);
    float nsq = 0.f;
    for (int i = 0; i < 64; i++) {
      int n = half * 64 + i;
      float t = (v[i] - mu) * rs * g_hn[h * 128 + n] + b_hn[h * 128 + n];
      if (t != t) t = 0.f;                           // nan_to_num
      else if (t ==  __builtin_inff()) t = 1.f;
      else if (t == -__builtin_inff()) t = -1.f;
      v[i] = t; nsq += t * t;
    }
    nsq += __shfl_xor(nsq, 1, 32);
    float inv = 1.f / fmaxf(sqrtf(nsq), 1e-12f);     // l2norm
    float t1 = 0.f, t2 = 0.f;
    for (int i = 0; i < 64; i++) { float t = v[i] * inv; v[i] = t; t1 += t; t2 += t * t; }
    t1 += __shfl_xor(t1, 1, 32); t2 += __shfl_xor(t2, 1, 32);
    float mu2 = t1 * (1.f / 128.f);
    float rs2 = rsqrtf(t2 * (1.f / 128.f) - mu2 * mu2 + 1e-5f);
    for (int i = 0; i < 64; i++) {
      int n = half * 64 + i;
      X1[r * P + n] = (bf16)((v[i] - mu2) * rs2 * g_r1[n] + b_r1[n]);
    }
  }
  __syncthreads();

  const int w = tid >> 5, lane = tid & 31, nn = lane & 15, hf = lane >> 4;
  {                                                  // 128->128 + GELU + LN
    v8f acc[8];
    for (int u = 0; u < 8; u++) acc[u] = zerov8();
    for (int k0 = 0; k0 < 128; k0 += 32) {
      v16bf af = load_a_frag(&X1[(w * 16) * P + k0], P);
      for (int u = 0; u < 8; u++)
        acc[u] = wmma_bf16(af, load_b_frag(Wr1 + (u * 16) * 128 + k0, 128), acc[u]);
    }
    float ge[8][8], rmu[8], rrs[8];
    for (int j = 0; j < 8; j++) {
      float s1 = 0.f, s2 = 0.f;
      for (int u = 0; u < 8; u++) {
        int n = u * 16 + nn;
        float x = acc[u][j] + bb_r1[n];
        float t = 0.5f * x * (1.0f + erff(x * 0.70710678118f));   // exact GELU
        ge[u][j] = t; s1 += t; s2 += t * t;
      }
      for (int msk = 1; msk < 16; msk <<= 1) {        // reduce across 16-lane half
        s1 += __shfl_xor(s1, msk, 32);
        s2 += __shfl_xor(s2, msk, 32);
      }
      float mu = s1 * (1.f / 128.f);
      rmu[j] = mu;
      rrs[j] = rsqrtf(s2 * (1.f / 128.f) - mu * mu + 1e-5f);
    }
    for (int u = 0; u < 8; u++)
      for (int j = 0; j < 8; j++) {
        int n = u * 16 + nn;
        int mm = w * 16 + hf * 8 + j;
        X2[mm * P + n] = (bf16)((ge[u][j] - rmu[j]) * rrs[j] * g_r2[n] + b_r2[n]);
      }
  }
  __syncthreads();

  {                                                  // 128->8 (padded to 16) + softmax
    v8f acc = zerov8();
    for (int k0 = 0; k0 < 128; k0 += 32) {
      v16bf af = load_a_frag(&X2[(w * 16) * P + k0], P);
      acc = wmma_bf16(af, load_b_frag(Wr2p + k0, 128), acc);
    }
    const bool valid = nn < 8;
    const float bb = valid ? bb_r2[nn] : 0.f;
    for (int j = 0; j < 8; j++) {
      float z = valid ? (acc[j] + bb) * 10.0f : -3.4e38f;   // /TEMP
      float mx = z;
      for (int msk = 1; msk < 8; msk <<= 1) mx = fmaxf(mx, __shfl_xor(mx, msk, 32));
      float e = expf(z - mx);
      float se = e;
      for (int msk = 1; msk < 8; msk <<= 1) se += __shfl_xor(se, msk, 32);
      float lg = z - mx - logf(se);
      float gate = fminf(fmaxf(expf(lg), 1e-6f), 1.0f);
      if (valid) {
        int b = w * 16 + hf * 8 + j;
        out[(((size_t)b * 12 + h) * 512 + s) * 8 + nn] = gate;
      }
    }
  }
}

// ---------------------------------------------------------------------------
// Launch: workspace layout (bytes)
//   [0, 201326592)        lin f32 (img @ 0, txt @ +100663296); reused as comb
//   [201326592, ...)      A_img bf16 [H][S][B][HD]   (50331648)
//   [251658240, ...)      A_txt bf16                 (50331648)
//   [301989888, ...)      W_ip bf16, W_tp bf16, 8 head weights bf16, W_r1, W_r2pad
// ---------------------------------------------------------------------------
extern "C" void kernel_launch(void* const* d_in, const int* in_sizes, int n_in,
                              void* d_out, int out_size, void* d_ws, size_t ws_size,
                              hipStream_t stream) {
  (void)in_sizes; (void)n_in; (void)out_size; (void)ws_size;
  const float* img  = (const float*)d_in[0];
  const float* txt  = (const float*)d_in[1];
  const float* W_ip = (const float*)d_in[2];
  const float* b_ip = (const float*)d_in[3];
  const float* g_ip = (const float*)d_in[4];
  const float* be_ip= (const float*)d_in[5];
  const float* W_tp = (const float*)d_in[6];
  const float* b_tp = (const float*)d_in[7];
  const float* g_tp = (const float*)d_in[8];
  const float* be_tp= (const float*)d_in[9];
  const float* g_hn = (const float*)d_in[26];
  const float* b_hn = (const float*)d_in[27];
  const float* g_r1 = (const float*)d_in[28];
  const float* b_r1 = (const float*)d_in[29];
  const float* W_r1 = (const float*)d_in[30];
  const float* bb_r1= (const float*)d_in[31];
  const float* g_r2 = (const float*)d_in[32];
  const float* b_r2 = (const float*)d_in[33];
  const float* W_r2 = (const float*)d_in[34];
  const float* bb_r2= (const float*)d_in[35];

  char* ws = (char*)d_ws;
  float* lin    = (float*)ws;                       // later aliased as comb
  bf16* Aimg    = (bf16*)(ws + 201326592);
  bf16* Atxt    = (bf16*)(ws + 251658240);
  bf16* Wip_b   = (bf16*)(ws + 301989888);
  bf16* Wtp_b   = (bf16*)(ws + 302776320);
  bf16* Whd_b   = (bf16*)(ws + 303562752);
  bf16* Wr1_b   = (bf16*)(ws + 304349184);
  bf16* Wr2p_b  = (bf16*)(ws + 304381952);

  cvt_f32_bf16<<<1536, 256, 0, stream>>>(W_ip, Wip_b, 393216);
  cvt_f32_bf16<<<1536, 256, 0, stream>>>(W_tp, Wtp_b, 393216);
  const int hsrc[8] = {10, 11, 12, 13, 18, 19, 20, 21};   // Wq/Wk/Wv/Wo x {i2t,t2i}
  for (int i = 0; i < 8; i++)
    cvt_f32_bf16<<<192, 256, 0, stream>>>((const float*)d_in[hsrc[i]],
                                          Whd_b + (size_t)i * 49152, 49152);
  cvt_f32_bf16<<<64, 256, 0, stream>>>(W_r1, Wr1_b, 16384);
  cvt_wr2_pad<<<8, 256, 0, stream>>>(W_r2, Wr2p_b);

  k1a_gemm<<<dim3(256, 6), 256, 0, stream>>>(img, Wip_b, lin);
  k1a_gemm<<<dim3(256, 6), 256, 0, stream>>>(txt, Wtp_b, lin + 25165824);

  k1b_ln<<<8192, 256, 0, stream>>>(lin, lin + 25165824,
                                   b_ip, g_ip, be_ip, b_tp, g_tp, be_tp, Aimg, Atxt);

  k2_attn<<<dim3(6144, 2), 128, 0, stream>>>(
      Aimg, Atxt, Whd_b,
      (const float*)d_in[14], (const float*)d_in[15],
      (const float*)d_in[16], (const float*)d_in[17],
      (const float*)d_in[22], (const float*)d_in[23],
      (const float*)d_in[24], (const float*)d_in[25],
      lin /* comb: reuses projection scratch, stream-ordered */);

  k3_router<<<6144, 128, 0, stream>>>(lin, g_hn, b_hn, g_r1, b_r1, Wr1_b, bb_r1,
                                      g_r2, b_r2, Wr2p_b, bb_r2, (float*)d_out);
}
